// ContrastiveMultiTaskLoss_48455821033481
// MI455X (gfx1250) — matrix operand order repaired and checked by reading, not compiled, not run on hardware
//
#include <hip/hip_runtime.h>
#include <hip/hip_bf16.h>

// ---------------------------------------------------------------------------
// ContrastiveMultiTaskLoss for MI455X (gfx1250, wave32, WMMA).
//
// loss = 1.0*MSE(price) + 0.5*MSE(change) + 0.3*BCEwithLogits(crit)
//      + 0.1 * NT-Xent(z1, z2, temp=0.1)
//
// NT-Xent: Z = [norm(z1); norm(z2)]  (16384 x 256)
//   loss_i = log(sum_{j != i} exp(<Z_i,Z_j>/T)) - <Z_i,Z_pos(i)>/T
// Fused Gram+LSE: sim in [-10,10] (unit vectors / 0.1) so plain sum-exp is
// numerically safe in f32 (max sum < 16384*e^10 ~ 3.6e8). One pass, no max.
//
// Fast path (ws >= ~8.4MB): pre-normalize Z into f16 once; hot loop is pure
// f16 loads + v_wmma_f32_16x16x32_f16 + exp epilogue (no converts, no norm
// scaling). 8MB f16 Z is L2-resident (192MB L2), so B-tile streams hit L2.
// Fallback path (small ws): on-the-fly f32->f16 convert, inv-norm epilogue.
// ---------------------------------------------------------------------------

typedef __attribute__((ext_vector_type(16))) _Float16 v16h;
typedef __attribute__((ext_vector_type(8)))  _Float16 v8h;
typedef __attribute__((ext_vector_type(8)))  float    v8f;
typedef __attribute__((ext_vector_type(4)))  float    v4f;

#define NROWS    8192
#define DIM      256
#define TROWS    16384          // 2*NROWS
#define NRTILES  (TROWS / 16)   // 1024 row tiles of 16
#define INV_TEMP 10.0f

__device__ __forceinline__ const float* zrow(const float* z1, const float* z2, int r) {
  return (r < NROWS) ? (z1 + (size_t)r * DIM) : (z2 + (size_t)(r - NROWS) * DIM);
}

// ===========================================================================
// FAST PATH
// ===========================================================================

// ---------------------------------------------------------------------------
// Kernel F1: fused row-norm + normalize + f32->f16 convert + store.
// One wave per row; lane owns a contiguous 8-float chunk (coalesced b128).
// ---------------------------------------------------------------------------
__global__ void normcvt_kernel(const float* __restrict__ z1, const float* __restrict__ z2,
                               _Float16* __restrict__ Zh) {
  const int gw   = (int)((blockIdx.x * blockDim.x + threadIdx.x) >> 5);
  const int lane = (int)(threadIdx.x & 31);
  if (gw >= TROWS) return;
  const float* rp = zrow(z1, z2, gw) + lane * 8;
  v4f a = ((const v4f*)rp)[0];
  v4f b = ((const v4f*)rp)[1];
  float s = a[0]*a[0] + a[1]*a[1] + a[2]*a[2] + a[3]*a[3]
          + b[0]*b[0] + b[1]*b[1] + b[2]*b[2] + b[3]*b[3];
#pragma unroll
  for (int m = 1; m < 32; m <<= 1) s += __shfl_xor(s, m, 32);
  const float inv = 1.0f / fmaxf(sqrtf(s), 1e-12f);
  v8h h;
#pragma unroll
  for (int i = 0; i < 4; ++i) {
    h[i]     = (_Float16)(a[i] * inv);
    h[4 + i] = (_Float16)(b[i] * inv);
  }
  *(v8h*)(Zh + (size_t)gw * DIM + lane * 8) = h;
}

// A-matrix 16x32 f16 tile (ISA 7.12.2): lane&15 = row M; lanes 0-15 hold
// K = {kb..kb+7, kb+16..kb+23}; lanes 16-31 hold K = {kb+8..kb+15, kb+24..kb+31}.
__device__ __forceinline__ v16h load_a16(const _Float16* rowp, int kb, int lane) {
  const int off = (lane >> 4) << 3;
  v8h x0 = *(const v8h*)(rowp + kb + off);
  v8h x1 = *(const v8h*)(rowp + kb + off + 16);
  return __builtin_shufflevector(x0, x1, 0,1,2,3,4,5,6,7,8,9,10,11,12,13,14,15);
}

// B-matrix 32x16 f16 tile (same K order as A):
// lane&15 = col N; lanes 0-15 hold K = {kb..kb+7, kb+16..kb+23}; lanes 16-31 K = {kb+8..kb+15, kb+24..kb+31}.
__device__ __forceinline__ v16h load_b16(const _Float16* colp, int kb, int lane) {
  const int off = (lane >> 4) << 3;
  v8h x0 = *(const v8h*)(colp + kb + off);
  v8h x1 = *(const v8h*)(colp + kb + off + 16);
  return __builtin_shufflevector(x0, x1, 0,1,2,3,4,5,6,7,8,9,10,11,12,13,14,15);
}

// ---------------------------------------------------------------------------
// Kernel F2: fused Gram-matmul + streaming sum-exp + positive extraction.
// One wave per 16-row tile; streams 1024 column tiles; 8 chained WMMAs each.
// Zh already normalized, so sim = c * INV_TEMP directly.
// ---------------------------------------------------------------------------
__global__ void ntxent_f16_kernel(const _Float16* __restrict__ Zh,
                                  float* __restrict__ row_partials) {
  const int wave = (int)((blockIdx.x * blockDim.x + threadIdx.x) >> 5);
  const int lane = (int)(threadIdx.x & 31);
  if (wave >= NRTILES) return;

  const int r0 = wave * 16;
  const _Float16* arow = Zh + (size_t)(r0 + (lane & 15)) * DIM;
  v16h A[8];
#pragma unroll
  for (int t = 0; t < 8; ++t) A[t] = load_a16(arow, 32 * t, lane);

  const int half8    = (lane >> 4) << 3;
  const int pos_base = (r0 < NROWS) ? (r0 + NROWS) : (r0 - NROWS);

  float sumexp[8], posv[8];
#pragma unroll
  for (int v = 0; v < 8; ++v) { sumexp[v] = 0.f; posv[v] = 0.f; }

  for (int cb = 0; cb < TROWS; cb += 16) {
    const _Float16* bcol = Zh + (size_t)(cb + (lane & 15)) * DIM;
    v8f c = {};
#pragma unroll
    for (int t = 0; t < 8; ++t) {
      v16h B = load_b16(bcol, 32 * t, lane);
      c = __builtin_amdgcn_wmma_f32_16x16x32_f16(false, A[t], false, B,
                                                 (short)0, c, false, false);
    }
    const bool diag_tile = (cb == r0);
    const bool pos_tile  = (cb == pos_base);
#pragma unroll
    for (int v = 0; v < 8; ++v) {
      // C layout: lane l, VGPR v -> tile_row = v + 8*(l>=16), tile_col = l&15.
      const bool dlane = ((lane & 15) == (v + half8));
      float s = c[v] * INV_TEMP;
      float e = __expf(s);
      if (diag_tile && dlane) e = 0.f;    // mask self-similarity
      sumexp[v] += e;
      if (pos_tile && dlane) posv[v] = s; // positive-pair logit
    }
  }

  float acc = 0.f;
#pragma unroll
  for (int v = 0; v < 8; ++v) {
    float se = sumexp[v], pv = posv[v];
#pragma unroll
    for (int m = 1; m < 16; m <<= 1) {
      se += __shfl_xor(se, m, 32);
      pv += __shfl_xor(pv, m, 32);
    }
    acc += __logf(se) - pv;
  }
  acc += __shfl_xor(acc, 16, 32);
  if (lane == 0) row_partials[wave] = acc;
}

// ===========================================================================
// FALLBACK PATH (small workspace): on-the-fly convert, deferred normalization.
// ===========================================================================

__device__ __forceinline__ v16h cvt16(const float* f) {
  v16h h;
#pragma unroll
  for (int i = 0; i < 16; ++i) h[i] = (_Float16)f[i];
  return h;
}

__device__ __forceinline__ v16h load_a_tile(const float* rowp, int kb, int lane) {
  const int off = (lane >> 4) << 3;
  const v4f* p0 = (const v4f*)(rowp + kb + off);
  const v4f* p1 = (const v4f*)(rowp + kb + off + 16);
  v4f x0 = p0[0], x1 = p0[1], x2 = p1[0], x3 = p1[1];
  float f[16];
#pragma unroll
  for (int i = 0; i < 4; ++i) { f[i] = x0[i]; f[4+i] = x1[i]; f[8+i] = x2[i]; f[12+i] = x3[i]; }
  return cvt16(f);
}

__device__ __forceinline__ v16h load_b_tile(const float* colp, int kb, int lane) {
  const int off = (lane >> 4) << 3;
  const v4f* p0 = (const v4f*)(colp + kb + off);
  const v4f* p1 = (const v4f*)(colp + kb + off + 16);
  v4f x0 = p0[0], x1 = p0[1], x2 = p1[0], x3 = p1[1];
  float f[16];
#pragma unroll
  for (int i = 0; i < 4; ++i) { f[i] = x0[i]; f[4+i] = x1[i]; f[8+i] = x2[i]; f[12+i] = x3[i]; }
  return cvt16(f);
}

__global__ void norm_kernel(const float* __restrict__ z1, const float* __restrict__ z2,
                            float* __restrict__ invn) {
  const int gw   = (int)((blockIdx.x * blockDim.x + threadIdx.x) >> 5);
  const int lane = (int)(threadIdx.x & 31);
  if (gw >= TROWS) return;
  const float* rp = zrow(z1, z2, gw);
  float s = 0.f;
#pragma unroll
  for (int k = 0; k < DIM / 32; ++k) { float v = rp[lane + 32 * k]; s += v * v; }
#pragma unroll
  for (int m = 1; m < 32; m <<= 1) s += __shfl_xor(s, m, 32);
  if (lane == 0) invn[gw] = 1.0f / fmaxf(sqrtf(s), 1e-12f);
}

__global__ void ntxent_kernel(const float* __restrict__ z1, const float* __restrict__ z2,
                              const float* __restrict__ invn, float* __restrict__ row_partials) {
  const int wave = (int)((blockIdx.x * blockDim.x + threadIdx.x) >> 5);
  const int lane = (int)(threadIdx.x & 31);
  if (wave >= NRTILES) return;

  const int r0 = wave * 16;
  const float* arow = zrow(z1, z2, r0 + (lane & 15));
  v16h A[8];
#pragma unroll
  for (int t = 0; t < 8; ++t) A[t] = load_a_tile(arow, 32 * t, lane);

  const int half8 = (lane >> 4) << 3;
  float invr[8];
#pragma unroll
  for (int v = 0; v < 8; ++v) invr[v] = invn[r0 + v + half8];

  const int pos_base = (r0 < NROWS) ? (r0 + NROWS) : (r0 - NROWS);
  float sumexp[8], posv[8];
#pragma unroll
  for (int v = 0; v < 8; ++v) { sumexp[v] = 0.f; posv[v] = 0.f; }

  for (int cb = 0; cb < TROWS; cb += 16) {
    const float* bcol = zrow(z1, z2, cb + (lane & 15));
    v8f c = {};
#pragma unroll
    for (int t = 0; t < 8; ++t) {
      v16h B = load_b_tile(bcol, 32 * t, lane);
      c = __builtin_amdgcn_wmma_f32_16x16x32_f16(false, A[t], false, B,
                                                 (short)0, c, false, false);
    }
    const float invc = invn[cb + (lane & 15)];
    const bool diag_tile = (cb == r0);
    const bool pos_tile  = (cb == pos_base);
#pragma unroll
    for (int v = 0; v < 8; ++v) {
      const bool dlane = ((lane & 15) == (v + half8));
      float s = c[v] * (invr[v] * invc) * INV_TEMP;
      float e = __expf(s);
      if (diag_tile && dlane) e = 0.f;
      sumexp[v] += e;
      if (pos_tile && dlane) posv[v] = s;
    }
  }

  float acc = 0.f;
#pragma unroll
  for (int v = 0; v < 8; ++v) {
    float se = sumexp[v], pv = posv[v];
#pragma unroll
    for (int m = 1; m < 16; m <<= 1) {
      se += __shfl_xor(se, m, 32);
      pv += __shfl_xor(pv, m, 32);
    }
    acc += __logf(se) - pv;
  }
  acc += __shfl_xor(acc, 16, 32);
  if (lane == 0) row_partials[wave] = acc;
}

// ===========================================================================
// Shared tail kernels
// ===========================================================================

__global__ void sup_kernel(const float* __restrict__ pp, const float* __restrict__ pt,
                           const float* __restrict__ cp, const float* __restrict__ ct,
                           const float* __restrict__ kp, const float* __restrict__ kt,
                           float* __restrict__ out3) {
  __shared__ float sA[256], sB[256], sC[256];
  const int tid = (int)threadIdx.x;
  float a = 0.f, b = 0.f, c = 0.f;
  for (int i = tid; i < NROWS; i += 256) {
    float dp = pp[i] - pt[i]; a += dp * dp;
    float dc = cp[i] - ct[i]; b += dc * dc;
    float x = kp[i], t = kt[i];
    c += fmaxf(x, 0.f) - x * t + log1pf(__expf(-fabsf(x)));
  }
  sA[tid] = a; sB[tid] = b; sC[tid] = c;
  __syncthreads();
  for (int s = 128; s > 0; s >>= 1) {
    if (tid < s) { sA[tid] += sA[tid + s]; sB[tid] += sB[tid + s]; sC[tid] += sC[tid + s]; }
    __syncthreads();
  }
  if (tid == 0) { out3[0] = sA[0]; out3[1] = sB[0]; out3[2] = sC[0]; }
}

__global__ void final_kernel(const float* __restrict__ row_partials,
                             const float* __restrict__ sup, float* __restrict__ out) {
  __shared__ float sh[256];
  const int tid = (int)threadIdx.x;
  float a = 0.f;
  for (int i = tid; i < NRTILES; i += 256) a += row_partials[i];
  sh[tid] = a;
  __syncthreads();
  for (int s = 128; s > 0; s >>= 1) {
    if (tid < s) sh[tid] += sh[tid + s];
    __syncthreads();
  }
  if (tid == 0) {
    const float ssl = sh[0] / (float)TROWS;
    const float supv = 1.0f * (sup[0] / (float)NROWS)
                     + 0.5f * (sup[1] / (float)NROWS)
                     + 0.3f * (sup[2] / (float)NROWS);
    out[0] = supv + 0.1f * ssl;
  }
}

// ---------------------------------------------------------------------------
extern "C" void kernel_launch(void* const* d_in, const int* in_sizes, int n_in,
                              void* d_out, int out_size, void* d_ws, size_t ws_size,
                              hipStream_t stream) {
  const float* price_pred  = (const float*)d_in[0];
  const float* price_tgt   = (const float*)d_in[1];
  const float* change_pred = (const float*)d_in[2];
  const float* change_tgt  = (const float*)d_in[3];
  const float* crit_pred   = (const float*)d_in[4];
  const float* crit_tgt    = (const float*)d_in[5];
  const float* z1          = (const float*)d_in[6];
  const float* z2          = (const float*)d_in[7];

  const size_t zh_bytes   = (size_t)TROWS * DIM * sizeof(_Float16);       // 8 MB
  const size_t tail_bytes = (size_t)(NRTILES + 16) * sizeof(float);
  const bool   fast       = (ws_size >= zh_bytes + tail_bytes);

  if (fast) {
    _Float16* Zh   = (_Float16*)d_ws;
    float*    tail = (float*)((char*)d_ws + zh_bytes);
    float*    rowp = tail;               // 1024 floats
    float*    sup3 = tail + NRTILES;     // 3 floats

    normcvt_kernel<<<TROWS / 8, 256, 0, stream>>>(z1, z2, Zh);
    sup_kernel<<<1, 256, 0, stream>>>(price_pred, price_tgt, change_pred, change_tgt,
                                      crit_pred, crit_tgt, sup3);
    ntxent_f16_kernel<<<NRTILES / 8, 256, 0, stream>>>(Zh, rowp);
    final_kernel<<<1, 256, 0, stream>>>(rowp, sup3, (float*)d_out);
  } else {
    float* ws   = (float*)d_ws;
    float* invn = ws;                    // 16384 floats
    float* rowp = ws + TROWS;            // 1024 floats
    float* sup3 = ws + TROWS + NRTILES;  // 3 floats (~70 KB total)

    norm_kernel<<<TROWS / 8, 256, 0, stream>>>(z1, z2, invn);
    sup_kernel<<<1, 256, 0, stream>>>(price_pred, price_tgt, change_pred, change_tgt,
                                      crit_pred, crit_tgt, sup3);
    ntxent_kernel<<<NRTILES / 8, 256, 0, stream>>>(z1, z2, invn, rowp);
    final_kernel<<<1, 256, 0, stream>>>(rowp, sup3, (float*)d_out);
  }
}
